// GATv2_49108656062977
// MI455X (gfx1250) — matrix operand order repaired and checked by reading, not compiled, not run on hardware
//
#include <hip/hip_runtime.h>
#include <hip/hip_bf16.h>
#include <math.h>

// ---------------------------------------------------------------------------
// GATv2 for MI455X (gfx1250, wave32).
//   N=50000 nodes, E=800000 edges, F_IN=256, H=4, C=64 (H*C=256).
// Stage 1: x@W_l+b_l, x@W_r+b_r via v_wmma_f32_16x16x32_bf16 (f32 accum).
// Stage 2: per-edge attention logits + segment max/sum + scatter aggregation
//          (memory-bound: ~3.5 GB irregular traffic -> ~150us @ 23.3 TB/s).
// ---------------------------------------------------------------------------

#define N_NODES 50000
#define N_EDGES 800000
#define F_IN    256
#define HC      256      // HEADS * C_OUT
#define HEADS   4
#define ET      (N_EDGES + N_NODES)   // edges + self loops
#define NEG_SLOPE 0.2f

typedef __attribute__((ext_vector_type(16))) __bf16 v16bf;
typedef __attribute__((ext_vector_type(8)))  float  v8f;

union Frag16 { v16bf v; unsigned int u[8]; };

// ---------------- init ----------------
__global__ void k_init_out(float* __restrict__ out, const float* __restrict__ bias) {
    int i = blockIdx.x * blockDim.x + threadIdx.x;
    if (i < N_NODES * HC) out[i] = bias[i & (HC - 1)];
}

__global__ void k_init_md(float* __restrict__ m, float* __restrict__ denom) {
    int i = blockIdx.x * blockDim.x + threadIdx.x;
    if (i < N_NODES * HEADS) { m[i] = -INFINITY; denom[i] = 0.0f; }
}

// ---------------- bf16 conversion ----------------
__global__ void k_cvt_x(const float* __restrict__ x, __bf16* __restrict__ xb) {
    int i = blockIdx.x * blockDim.x + threadIdx.x;
    if (i < N_NODES * F_IN) xb[i] = (__bf16)x[i];
}

// W is [K=256][N=256] row-major; store transposed [N][K] so B-fragment K-pairs
// are contiguous dwords.
__global__ void k_cvt_w(const float* __restrict__ W, __bf16* __restrict__ Wt) {
    int i = blockIdx.x * blockDim.x + threadIdx.x;   // 65536
    if (i < F_IN * HC) {
        int k = i >> 8, n = i & 255;
        Wt[n * F_IN + k] = (__bf16)W[i];
    }
}

// ---------------- WMMA GEMM: out = xb @ W + b  (M=50000, N=256, K=256) ------
// Block = 256 threads = 8 waves. Each wave computes two 16x16 N-tiles of one
// 16-row M-tile. K loop: 8 steps of K=32. blockIdx.y selects (W_l)/(W_r).
__global__ void __launch_bounds__(256)
k_gemm(const __bf16* __restrict__ xb,
       const __bf16* __restrict__ wtl, const __bf16* __restrict__ wtr,
       const float* __restrict__ bl,   const float* __restrict__ br,
       float* __restrict__ xl,         float* __restrict__ xr) {
    const __bf16* W   = blockIdx.y ? wtr : wtl;
    const float*  bia = blockIdx.y ? br  : bl;
    float*        out = blockIdx.y ? xr  : xl;

    const int wave = threadIdx.x >> 5;      // 0..7
    const int lane = threadIdx.x & 31;
    const int half = lane >> 4;             // 0/1
    const int ln   = lane & 15;
    const int M0   = blockIdx.x * 16;       // 3125 tiles, exact
    const int row  = M0 + ln;               // A row this lane supplies
    const int n0a  = (2 * wave) * 16;
    const int n0b  = n0a + 16;

    v8f acc0 = {};
    v8f acc1 = {};

    const __bf16* arow = xb + (size_t)row * F_IN;
    const __bf16* b0p  = W + (size_t)(n0a + ln) * F_IN;
    const __bf16* b1p  = W + (size_t)(n0b + ln) * F_IN;

#pragma unroll
    for (int kb = 0; kb < 8; ++kb) {
        const int K0 = kb * 32;
        Frag16 a, b0, b1;
#pragma unroll
        for (int v = 0; v < 8; ++v) {
            // ISA 7.12.2: 16-bit A 16x32 — lanes 0-15 hold K {0..7,16..23},
            // lanes 16-31 hold K {8..15,24..31}, packed 2 per VGPR.
            const int ka = ((v < 4) ? 2 * v : 16 + 2 * (v - 4)) + 8 * half;
            a.u[v] = *(const unsigned int*)(arow + K0 + ka);
            // B 32x16: same K order as A — lanes 0-15 -> K {0..7,16..23}, lanes 16-31 -> K {8..15,24..31} (col = ln)
            const int kbo = ((v < 4) ? 2 * v : 16 + 2 * (v - 4)) + 8 * half;
            b0.u[v] = *(const unsigned int*)(b0p + K0 + kbo);
            b1.u[v] = *(const unsigned int*)(b1p + K0 + kbo);
        }
        acc0 = __builtin_amdgcn_wmma_f32_16x16x32_bf16(false, a.v, false, b0.v,
                                                       (short)0, acc0, false, false);
        acc1 = __builtin_amdgcn_wmma_f32_16x16x32_bf16(false, a.v, false, b1.v,
                                                       (short)0, acc1, false, false);
    }

    // C/D layout: VGPR v, lanes 0-15 -> (M=v, N=lane); lanes 16-31 -> (M=8+v).
#pragma unroll
    for (int v = 0; v < 8; ++v) {
        const int m = M0 + v + 8 * half;
        out[(size_t)m * HC + n0a + ln] = acc0[v] + bia[n0a + ln];
        out[(size_t)m * HC + n0b + ln] = acc1[v] + bia[n0b + ln];
    }
}

// ---------------- edge pass 1: logits + segment max ----------------
__device__ inline void atomicMaxF(float* addr, float v) {
    if (v >= 0.0f) atomicMax((int*)addr, __float_as_int(v));
    else           atomicMin((unsigned int*)addr, __float_as_uint(v));
}

// One wave per edge; lane handles 8 consecutive channels (head = lane>>3).
__global__ void __launch_bounds__(256)
k_logits(const int* __restrict__ ei,
         const float* __restrict__ xl, const float* __restrict__ xr,
         const float* __restrict__ att,
         float* __restrict__ logits, float* __restrict__ m) {
    const int wave = threadIdx.x >> 5;
    const int lane = threadIdx.x & 31;
    const long long e = (long long)blockIdx.x * 8 + wave;
    if (e >= ET) return;
    const int src = (e < N_EDGES) ? ei[e]           : (int)(e - N_EDGES);
    const int dst = (e < N_EDGES) ? ei[N_EDGES + e] : (int)(e - N_EDGES);

    const int c0 = lane * 8;
    const float4 l0 = *(const float4*)(xl + (size_t)src * HC + c0);
    const float4 l1 = *(const float4*)(xl + (size_t)src * HC + c0 + 4);
    const float4 r0 = *(const float4*)(xr + (size_t)dst * HC + c0);
    const float4 r1 = *(const float4*)(xr + (size_t)dst * HC + c0 + 4);
    const float4 a0 = *(const float4*)(att + c0);
    const float4 a1 = *(const float4*)(att + c0 + 4);

    float s[8] = { l0.x + r0.x, l0.y + r0.y, l0.z + r0.z, l0.w + r0.w,
                   l1.x + r1.x, l1.y + r1.y, l1.z + r1.z, l1.w + r1.w };
    const float aa[8] = { a0.x, a0.y, a0.z, a0.w, a1.x, a1.y, a1.z, a1.w };
    float p = 0.0f;
#pragma unroll
    for (int j = 0; j < 8; ++j) {
        float v = s[j];
        v = (v > 0.0f) ? v : v * NEG_SLOPE;   // LeakyReLU
        p += v * aa[j];
    }
    // reduce across the 8 lanes of this head
    p += __shfl_xor(p, 1, 32);
    p += __shfl_xor(p, 2, 32);
    p += __shfl_xor(p, 4, 32);
    if ((lane & 7) == 0) {
        const int h = lane >> 3;
        logits[e * HEADS + h] = p;
        atomicMaxF(&m[(size_t)dst * HEADS + h], p);
    }
}

// ---------------- edge pass 2: denom = segment_sum(exp(logit - m)) ----------
__global__ void k_denom(const int* __restrict__ ei,
                        const float* __restrict__ logits,
                        const float* __restrict__ m,
                        float* __restrict__ denom) {
    const long long i = (long long)blockIdx.x * blockDim.x + threadIdx.x;
    if (i >= (long long)ET * HEADS) return;
    const long long e = i >> 2;
    const int h = (int)(i & 3);
    const int dst = (e < N_EDGES) ? ei[N_EDGES + e] : (int)(e - N_EDGES);
    const float a = __expf(logits[i] - m[(size_t)dst * HEADS + h]);
    atomicAdd(&denom[(size_t)dst * HEADS + h], a);
}

// ---------------- edge pass 3: out[dst] += alpha * x_l[src] ----------------
__global__ void __launch_bounds__(256)
k_aggregate(const int* __restrict__ ei,
            const float* __restrict__ xl,
            const float* __restrict__ logits,
            const float* __restrict__ m, const float* __restrict__ denom,
            float* __restrict__ out) {
    const int wave = threadIdx.x >> 5;
    const int lane = threadIdx.x & 31;
    const long long e = (long long)blockIdx.x * 8 + wave;
    if (e >= ET) return;
    const int src = (e < N_EDGES) ? ei[e]           : (int)(e - N_EDGES);
    const int dst = (e < N_EDGES) ? ei[N_EDGES + e] : (int)(e - N_EDGES);

    const int h = lane >> 3;
    const float lg = logits[e * HEADS + h];
    const float mv = m[(size_t)dst * HEADS + h];
    const float dn = denom[(size_t)dst * HEADS + h];
    const float alpha = __expf(lg - mv) / dn;

    const int c0 = lane * 8;
    const float4 s0 = *(const float4*)(xl + (size_t)src * HC + c0);
    const float4 s1 = *(const float4*)(xl + (size_t)src * HC + c0 + 4);
    float* o = out + (size_t)dst * HC + c0;
    atomicAdd(o + 0, alpha * s0.x);
    atomicAdd(o + 1, alpha * s0.y);
    atomicAdd(o + 2, alpha * s0.z);
    atomicAdd(o + 3, alpha * s0.w);
    atomicAdd(o + 4, alpha * s1.x);
    atomicAdd(o + 5, alpha * s1.y);
    atomicAdd(o + 6, alpha * s1.z);
    atomicAdd(o + 7, alpha * s1.w);
}

// ---------------------------------------------------------------------------
extern "C" void kernel_launch(void* const* d_in, const int* in_sizes, int n_in,
                              void* d_out, int out_size, void* d_ws, size_t ws_size,
                              hipStream_t stream) {
    const float* x    = (const float*)d_in[0];
    const int*   ei   = (const int*)  d_in[1];   // [2, E] int32 (jax default x64-off)
    const float* W_l  = (const float*)d_in[2];
    const float* b_l  = (const float*)d_in[3];
    const float* W_r  = (const float*)d_in[4];
    const float* b_r  = (const float*)d_in[5];
    const float* att  = (const float*)d_in[6];   // [H, C] flat == per-channel
    const float* bias = (const float*)d_in[7];
    float* out = (float*)d_out;

    // workspace carve-up (all offsets multiples of 256B)
    char* ws = (char*)d_ws;
    size_t off = 0;
    __bf16* xb  = (__bf16*)(ws + off); off += (size_t)N_NODES * F_IN * 2;   // 25.6 MB
    __bf16* wtl = (__bf16*)(ws + off); off += (size_t)F_IN * HC * 2;        // 128 KB
    __bf16* wtr = (__bf16*)(ws + off); off += (size_t)F_IN * HC * 2;        // 128 KB
    float*  xl  = (float*) (ws + off); off += (size_t)N_NODES * HC * 4;     // 51.2 MB
    float*  xr  = (float*) (ws + off); off += (size_t)N_NODES * HC * 4;     // 51.2 MB
    float*  lg  = (float*) (ws + off); off += (size_t)ET * HEADS * 4;       // 13.6 MB
    float*  mb  = (float*) (ws + off); off += (size_t)N_NODES * HEADS * 4;  // 0.8 MB
    float*  dnb = (float*) (ws + off); off += (size_t)N_NODES * HEADS * 4;  // 0.8 MB

    // init
    k_init_out<<<(N_NODES * HC + 255) / 256, 256, 0, stream>>>(out, bias);
    k_init_md <<<(N_NODES * HEADS + 255) / 256, 256, 0, stream>>>(mb, dnb);

    // bf16 conversion + weight transpose
    k_cvt_x<<<(N_NODES * F_IN + 255) / 256, 256, 0, stream>>>(x, xb);
    k_cvt_w<<<(F_IN * HC + 255) / 256, 256, 0, stream>>>(W_l, wtl);
    k_cvt_w<<<(F_IN * HC + 255) / 256, 256, 0, stream>>>(W_r, wtr);

    // WMMA GEMMs: grid.y = {W_l, W_r}
    dim3 ggrid(N_NODES / 16, 2);
    k_gemm<<<ggrid, 256, 0, stream>>>(xb, wtl, wtr, b_l, b_r, xl, xr);

    // edge stages
    const int eblocks = (ET + 7) / 8;
    k_logits   <<<eblocks, 256, 0, stream>>>(ei, xl, xr, att, lg, mb);
    k_denom    <<<(int)(((long long)ET * HEADS + 255) / 256), 256, 0, stream>>>(ei, lg, mb, dnb);
    k_aggregate<<<eblocks, 256, 0, stream>>>(ei, xl, lg, mb, dnb, out);
}
